// HybridTransformerESN_12360915878153
// MI455X (gfx1250) — matrix operand order repaired and checked by reading, not compiled, not run on hardware
//
#include <hip/hip_runtime.h>

// ---------------------------------------------------------------------------
// HybridTransformerESN for MI455X (gfx1250, wave32, WMMA).
//   B=16, S=512, IN=32, D=512, H=8, HD=64, L=4, FF=2048, R=1024
// Strategy: bf16-in / f32-accumulate WMMA GEMMs (v_wmma_f32_16x16x32_bf16)
// for every matmul; wave-per-row softmax/layernorm; persistent single-WG
// ESN scan with state double-buffered in LDS (WGP has 320KB) and bf16 W_res
// streamed from L2 each step.
//
// Round-1 fix: keep the scan's K loop rolled (partial unroll) so the compiler
// cannot hoist the t-invariant W_res fragment loads out of the 512-step
// recurrence (previously it fully unrolled, hoisted ~512 VGPRs of fragments,
// and spilled them to scratch -> scratch_load_b128 reloads in the hot loop).
// ---------------------------------------------------------------------------

typedef __attribute__((ext_vector_type(16))) __bf16          v16bf;
typedef __attribute__((ext_vector_type(8)))  float           v8f;
typedef __attribute__((ext_vector_type(8)))  unsigned short  u8s;
typedef __attribute__((ext_vector_type(16))) unsigned short  u16s;
typedef __attribute__((ext_vector_type(4)))  float           f4v;

__device__ inline unsigned short f32_to_bf16(float f) {
  unsigned u = __builtin_bit_cast(unsigned, f);
  unsigned r = 0x7FFFu + ((u >> 16) & 1u);           // round-to-nearest-even
  return (unsigned short)((u + r) >> 16);
}
__device__ inline float bf16_to_f32(unsigned short u) {
  unsigned x = ((unsigned)u) << 16;
  return __builtin_bit_cast(float, x);
}
__device__ inline v16bf pack16(u8s lo, u8s hi) {
  u16s u = __builtin_shufflevector(lo, hi, 0,1,2,3,4,5,6,7,8,9,10,11,12,13,14,15);
  return __builtin_bit_cast(v16bf, u);
}

// A fragment (16x32 bf16, M x K) from LDS tile laid out [row][k], rowStride in
// ushorts (multiple of 8). Lane l: row = mbase+(l&15); K = {0..15} + 16*(l>=16) (contiguous).
__device__ inline v16bf load_frag_a_lds(const unsigned short* tile, int rowStride,
                                        int mbase, int kOff, int lane) {
  int row = mbase + (lane & 15);
  int off = (lane & 16) ? 16 : 0;
  const unsigned short* p = tile + row * rowStride + kOff + off;
  u8s lo = *(const u8s*)(p);
  u8s hi = *(const u8s*)(p + 8);
  return pack16(lo, hi);
}
// B fragment (32x16 bf16, K x N) from LDS tile laid out [n][k].
// Lane l: col = nbase+(l&15); K = {0..15} + 16*(l>=16) (contiguous).
__device__ inline v16bf load_frag_b_lds(const unsigned short* tile, int rowStride,
                                        int nbase, int lane) {
  int col = nbase + (lane & 15);
  int koff = (lane & 16) ? 16 : 0;
  const unsigned short* p = tile + col * rowStride + koff;
  u8s lo = *(const u8s*)(p);
  u8s hi = *(const u8s*)(p + 8);
  return pack16(lo, hi);
}
// B fragment directly from a row-major bf16 matrix in global memory ([N][K], ld
// in elements): used by the ESN scan (W_res stays in L2).
__device__ inline v16bf load_frag_b_glb(const unsigned short* W, int ld,
                                        int nbase, int kOff, int lane) {
  int col = nbase + (lane & 15);
  int koff = kOff + ((lane & 16) ? 16 : 0);
  const unsigned short* p = W + (long long)col * ld + koff;
  u8s lo = *(const u8s*)(p);
  u8s hi = *(const u8s*)(p + 8);
  return pack16(lo, hi);
}

__device__ inline v8f wmma_bf16(v16bf a, v16bf b, v8f c) {
  return __builtin_amdgcn_wmma_f32_16x16x32_bf16(false, a, false, b,
                                                 (short)0, c, false, false);
}

// ---------------------------------------------------------------------------
// Generic batched GEMM:  C[z] = act(alpha * A[z] @ opB(B[z]) + bias [+ res])
//   A: [M,K] f32 (row stride lda).  B: [N,K] (B_KxN=false) or [K,N] (true).
//   Block tile 128(M) x 64(N), 8 waves (4x2), wave tile 32x32, K-chunk 32.
//   z offsets: base += (z/zdiv)*s?0 + (z%zdiv)*s?1.
// ---------------------------------------------------------------------------
#define LDSP 40   // padded K-stride in ushorts (multiple of 8; 20 dwords -> no bank conflicts)

template <bool B_KxN>
__global__ __launch_bounds__(256) void gemm_bf16_wmma(
    const float* __restrict__ A, int lda,
    const float* __restrict__ B, int ldb,
    float* __restrict__ C, int ldc,
    const float* __restrict__ bias,
    const float* __restrict__ res, int ldres,
    int M, int N, int K, float alpha, int relu, int zdiv,
    long long sA0, long long sA1, long long sB0, long long sB1,
    long long sC0, long long sC1) {
  __shared__ unsigned short As[128 * LDSP];
  __shared__ unsigned short Bs[64 * LDSP];

  const int tid  = threadIdx.x;
  const int lane = tid & 31;
  const int w    = tid >> 5;   // 8 waves
  const int wm   = w & 3;      // 4 waves along M
  const int wn   = w >> 2;     // 2 waves along N
  const int m0   = blockIdx.y * 128;
  const int n0   = blockIdx.x * 64;

  const long long zq = blockIdx.z / zdiv, zr = blockIdx.z % zdiv;
  A += zq * sA0 + zr * sA1;
  B += zq * sB0 + zr * sB1;
  C += zq * sC0 + zr * sC1;
  if (res) res += zq * sC0 + zr * sC1;

  v8f acc[2][2] = {};

  for (int k0 = 0; k0 < K; k0 += 32) {
    // --- stage A (128x32 f32 -> bf16 LDS), 8 threads/row x float4, 4 passes
    {
      int r = tid >> 3;
      int c = (tid & 7) * 4;
#pragma unroll
      for (int p = 0; p < 4; ++p) {
        int row = r + p * 32;
        f4v v = *(const f4v*)(A + (long long)(m0 + row) * lda + k0 + c);
        unsigned short* d = &As[row * LDSP + c];
        d[0] = f32_to_bf16(v.x); d[1] = f32_to_bf16(v.y);
        d[2] = f32_to_bf16(v.z); d[3] = f32_to_bf16(v.w);
      }
    }
    // --- stage B (64x32) into canonical [n][k] layout
    if (!B_KxN) {
      int r = tid >> 3;
      int c = (tid & 7) * 4;
#pragma unroll
      for (int p = 0; p < 2; ++p) {
        int n = r + p * 32;
        f4v v = *(const f4v*)(B + (long long)(n0 + n) * ldb + k0 + c);
        unsigned short* d = &Bs[n * LDSP + c];
        d[0] = f32_to_bf16(v.x); d[1] = f32_to_bf16(v.y);
        d[2] = f32_to_bf16(v.z); d[3] = f32_to_bf16(v.w);
      }
    } else {
      int kk = tid >> 4;
      int nn = (tid & 15) * 4;
#pragma unroll
      for (int p = 0; p < 2; ++p) {
        int k = kk + p * 16;
        f4v v = *(const f4v*)(B + (long long)(k0 + k) * ldb + n0 + nn);
        Bs[(nn + 0) * LDSP + k] = f32_to_bf16(v.x);
        Bs[(nn + 1) * LDSP + k] = f32_to_bf16(v.y);
        Bs[(nn + 2) * LDSP + k] = f32_to_bf16(v.z);
        Bs[(nn + 3) * LDSP + k] = f32_to_bf16(v.w);
      }
    }
    __syncthreads();

    v16bf af0 = load_frag_a_lds(As, LDSP, wm * 32 + 0,  0, lane);
    v16bf af1 = load_frag_a_lds(As, LDSP, wm * 32 + 16, 0, lane);
    v16bf bf0 = load_frag_b_lds(Bs, LDSP, wn * 32 + 0,  lane);
    v16bf bf1 = load_frag_b_lds(Bs, LDSP, wn * 32 + 16, lane);
    acc[0][0] = wmma_bf16(af0, bf0, acc[0][0]);
    acc[0][1] = wmma_bf16(af0, bf1, acc[0][1]);
    acc[1][0] = wmma_bf16(af1, bf0, acc[1][0]);
    acc[1][1] = wmma_bf16(af1, bf1, acc[1][1]);
    __syncthreads();
  }

  // --- epilogue: C layout per ISA (VGPR v, lane l -> m = v+8*(l>=16), n = l&15)
#pragma unroll
  for (int i = 0; i < 2; ++i) {
    int mBase = m0 + wm * 32 + i * 16 + ((lane & 16) ? 8 : 0);
#pragma unroll
    for (int j = 0; j < 2; ++j) {
      int n = n0 + wn * 32 + j * 16 + (lane & 15);
      float bi = bias ? bias[n] : 0.0f;
#pragma unroll
      for (int v = 0; v < 8; ++v) {
        int m = mBase + v;
        float val = acc[i][j][v] * alpha + bi;
        if (res) val += res[(long long)m * ldres + n];
        if (relu) val = fmaxf(val, 0.0f);
        C[(long long)m * ldc + n] = val;
      }
    }
  }
}

// ---------------------------------------------------------------------------
// LayerNorm over last dim D=512: one wave per row (wave32).
// ---------------------------------------------------------------------------
__global__ __launch_bounds__(256) void layernorm_k(
    const float* __restrict__ X, const float* __restrict__ g,
    const float* __restrict__ b, float* __restrict__ Y, int D, int rows) {
  int row  = blockIdx.x * 8 + (threadIdx.x >> 5);
  int lane = threadIdx.x & 31;
  if (row >= rows) return;
  const float* x = X + (long long)row * D;
  float sum = 0.f, sq = 0.f;
  for (int i = lane; i < D; i += 32) { float v = x[i]; sum += v; sq += v * v; }
#pragma unroll
  for (int o = 16; o > 0; o >>= 1) {
    sum += __shfl_down(sum, o, 32);
    sq  += __shfl_down(sq,  o, 32);
  }
  sum = __shfl(sum, 0, 32); sq = __shfl(sq, 0, 32);
  float mu  = sum / D;
  float var = sq / D - mu * mu;
  float inv = rsqrtf(var + 1e-5f);
  float* y = Y + (long long)row * D;
  for (int i = lane; i < D; i += 32) y[i] = (x[i] - mu) * inv * g[i] + b[i];
}

// ---------------------------------------------------------------------------
// Softmax over rows of length Ncols: one wave per row.
// ---------------------------------------------------------------------------
__global__ __launch_bounds__(256) void softmax_k(float* __restrict__ S, int Ncols,
                                                 long long rows) {
  long long row = (long long)blockIdx.x * 8 + (threadIdx.x >> 5);
  int lane = threadIdx.x & 31;
  if (row >= rows) return;
  float* p = S + row * Ncols;
  float mx = -1e30f;
  for (int i = lane; i < Ncols; i += 32) mx = fmaxf(mx, p[i]);
#pragma unroll
  for (int o = 16; o > 0; o >>= 1) mx = fmaxf(mx, __shfl_down(mx, o, 32));
  mx = __shfl(mx, 0, 32);
  float s = 0.f;
  for (int i = lane; i < Ncols; i += 32) { float e = __expf(p[i] - mx); p[i] = e; s += e; }
#pragma unroll
  for (int o = 16; o > 0; o >>= 1) s += __shfl_down(s, o, 32);
  s = __shfl(s, 0, 32);
  float inv = 1.0f / s;
  for (int i = lane; i < Ncols; i += 32) p[i] *= inv;
}

// f32 -> bf16 bulk convert (W_res, done once; scan re-reads it 512x from L2).
__global__ void cvt_bf16_k(const float* __restrict__ X,
                           unsigned short* __restrict__ Y, long long n) {
  long long i = (long long)blockIdx.x * 256 + threadIdx.x;
  if (i < n) Y[i] = f32_to_bf16(X[i]);
}

// ---------------------------------------------------------------------------
// ESN scan: single persistent workgroup, 1024 threads = 32 waves on one WGP.
// state[16][1024] double-buffered in LDS (bf16). Each step:
//   acc = state_bf16 @ W_res^T (WMMA, wave w owns columns w*32..w*32+31)
//   state' = tanh(acc + u[:,t,:]); one s_barrier per step.
// Finally: out[b] = state[b,:] . ro_W + ro_b  (wave-per-batch reduction).
// ---------------------------------------------------------------------------
#define RST 1032  // 1024 + 8 pad (ushorts); 516 dwords/row -> conflict-free A reads

__global__ __launch_bounds__(1024) void esn_scan_k(
    const float* __restrict__ u,              // [B=16, S=512, R=1024] f32
    const unsigned short* __restrict__ Wres,  // [R, R] bf16 row-major
    const float* __restrict__ roW, const float* __restrict__ rob,
    float* __restrict__ out) {                // [16]
  __shared__ unsigned short st[2][16 * RST];
  const int tid  = threadIdx.x;
  const int lane = tid & 31;
  const int w    = tid >> 5;      // 0..31
  const int nb   = w * 32;        // this wave's output-column base

  for (int i = tid; i < 16 * RST; i += 1024) { st[0][i] = 0; st[1][i] = 0; }
  __syncthreads();

  int buf = 0;
  for (int t = 0; t < 512; ++t) {
    // Compiler barrier: forbid hoisting the t-invariant W_res fragment loads
    // out of the recurrence (they don't fit in VGPRs and would spill).
    asm volatile("" ::: "memory");
    v8f acc0 = {}, acc1 = {};
    const unsigned short* sb = st[buf];
    // Keep rolled (factor 4): addresses stay induction-dependent -> no LICM,
    // no spills; ~24 transient VGPRs of fragments in flight.
#pragma unroll 4
    for (int kc = 0; kc < 1024; kc += 32) {
      v16bf a  = load_frag_a_lds(sb, RST, 0, kc, lane);     // 16 batches x 32 K
      v16bf b0 = load_frag_b_glb(Wres, 1024, nb,      kc, lane);
      v16bf b1 = load_frag_b_glb(Wres, 1024, nb + 16, kc, lane);
      acc0 = wmma_bf16(a, b0, acc0);
      acc1 = wmma_bf16(a, b1, acc1);
    }
    unsigned short* sn = st[buf ^ 1];
    int mOff = (lane & 16) ? 8 : 0;
    int n0 = nb + (lane & 15);
    int n1 = nb + 16 + (lane & 15);
#pragma unroll
    for (int v = 0; v < 8; ++v) {
      int b = mOff + v;
      long long ub = ((long long)b * 512 + t) * 1024;
      float s0 = tanhf(acc0[v] + u[ub + n0]);
      float s1 = tanhf(acc1[v] + u[ub + n1]);
      sn[b * RST + n0] = f32_to_bf16(s0);
      sn[b * RST + n1] = f32_to_bf16(s1);
    }
    __syncthreads();
    buf ^= 1;
  }

  if (w < 16) {   // wave w reduces batch b = w
    const unsigned short* sf = st[buf] + w * RST;
    float s = 0.f;
    for (int r = lane; r < 1024; r += 32) s += bf16_to_f32(sf[r]) * roW[r];
#pragma unroll
    for (int o = 16; o > 0; o >>= 1) s += __shfl_down(s, o, 32);
    if (lane == 0) out[w] = s + rob[0];
  }
}

// ---------------------------------------------------------------------------
// Host-side launcher.
// ---------------------------------------------------------------------------
static inline void launch_gemm(bool bkxn, dim3 grid, const float* A, int lda,
                               const float* B, int ldb, float* C, int ldc,
                               const float* bias, const float* res, int ldres,
                               int M, int N, int K, float alpha, int relu,
                               int zdiv, long long sA0, long long sA1,
                               long long sB0, long long sB1, long long sC0,
                               long long sC1, hipStream_t st) {
  if (bkxn)
    gemm_bf16_wmma<true><<<grid, 256, 0, st>>>(A, lda, B, ldb, C, ldc, bias, res,
        ldres, M, N, K, alpha, relu, zdiv, sA0, sA1, sB0, sB1, sC0, sC1);
  else
    gemm_bf16_wmma<false><<<grid, 256, 0, st>>>(A, lda, B, ldb, C, ldc, bias, res,
        ldres, M, N, K, alpha, relu, zdiv, sA0, sA1, sB0, sB1, sC0, sC1);
}

extern "C" void kernel_launch(void* const* d_in, const int* in_sizes, int n_in,
                              void* d_out, int out_size, void* d_ws, size_t ws_size,
                              hipStream_t stream) {
  (void)in_sizes; (void)n_in; (void)out_size; (void)ws_size;
  // setup_inputs() dict order (W_res appended last):
  const float* x      = (const float*)d_in[0];   // [16,512,32]
  const float* proj_W = (const float*)d_in[1];   // [512,32]
  const float* proj_b = (const float*)d_in[2];   // [512]
  const float* qkv_W  = (const float*)d_in[3];   // [4,1536,512]
  const float* qkv_b  = (const float*)d_in[4];   // [4,1536]
  const float* out_W  = (const float*)d_in[5];   // [4,512,512]
  const float* out_b  = (const float*)d_in[6];   // [4,512]
  const float* ln1_g  = (const float*)d_in[7];   // [4,512]
  const float* ln1_b  = (const float*)d_in[8];
  const float* ff1_W  = (const float*)d_in[9];   // [4,2048,512]
  const float* ff1_b  = (const float*)d_in[10];  // [4,2048]
  const float* ff2_W  = (const float*)d_in[11];  // [4,512,2048]
  const float* ff2_b  = (const float*)d_in[12];  // [4,512]
  const float* ln2_g  = (const float*)d_in[13];
  const float* ln2_b  = (const float*)d_in[14];
  const float* W_in   = (const float*)d_in[15];  // [1024,512]
  const float* ro_W   = (const float*)d_in[16];  // [1,1024]
  const float* ro_b   = (const float*)d_in[17];  // [1]
  const float* W_res  = (const float*)d_in[18];  // [1024,1024]

  const int Bsz = 16, S = 512, IN = 32, D = 512, H = 8, HD = 64, L = 4,
            FF = 2048, R = 1024;
  const int M = Bsz * S;  // 8192 token rows

  // workspace carve-up (f32 unless noted), ~322 MiB total
  char* p = (char*)d_ws;
  float* h0     = (float*)p; p += (size_t)M * D * 4;        // current hidden
  float* t1     = (float*)p; p += (size_t)M * D * 4;        // pre-LN temp
  float* attn_o = (float*)p; p += (size_t)M * D * 4;        // attention output
  float* qkv    = (float*)p; p += (size_t)M * 3 * D * 4;    // [B,S,3D]
  float* ff1o   = (float*)p; p += (size_t)M * FF * 4;       // [B,S,FF]
  float* uu     = (float*)p; p += (size_t)M * R * 4;        // ESN drive [B,S,R]
  float* scores = (float*)p; p += (size_t)Bsz * H * S * S * 4;  // [B,H,S,S]
  unsigned short* wres_bf = (unsigned short*)p;             // [R,R] bf16

  // 1) input projection: h0 = x @ proj_W^T + proj_b    (M=8192,N=512,K=32)
  launch_gemm(false, dim3(D / 64, M / 128, 1), x, IN, proj_W, IN, h0, D,
              proj_b, nullptr, 0, M, D, IN, 1.f, 0, 1, 0, 0, 0, 0, 0, 0, stream);

  for (int l = 0; l < L; ++l) {
    const float* Wqkv = qkv_W + (size_t)l * 3 * D * D;
    const float* bqkv = qkv_b + (size_t)l * 3 * D;
    const float* Wo   = out_W + (size_t)l * D * D;
    const float* bo   = out_b + (size_t)l * D;
    const float* W1   = ff1_W + (size_t)l * FF * D;
    const float* b1   = ff1_b + (size_t)l * FF;
    const float* W2   = ff2_W + (size_t)l * D * FF;
    const float* b2   = ff2_b + (size_t)l * D;

    // 2a) qkv = h0 @ Wqkv^T + bqkv     (N=1536,K=512)
    launch_gemm(false, dim3(3 * D / 64, M / 128, 1), h0, D, Wqkv, D, qkv, 3 * D,
                bqkv, nullptr, 0, M, 3 * D, D, 1.f, 0, 1, 0, 0, 0, 0, 0, 0, stream);

    // 2b) scores[b,h] = (q @ k^T) / sqrt(HD)   batched z = b*H+h (128)
    launch_gemm(false, dim3(S / 64, S / 128, Bsz * H),
                qkv /*q*/, 3 * D, qkv + D /*k*/, 3 * D, scores, S,
                nullptr, nullptr, 0, S, S, HD, 0.125f, 0, H,
                (long long)S * 3 * D, HD,            // A strides (b, h)
                (long long)S * 3 * D, HD,            // B strides
                (long long)H * S * S, (long long)S * S, stream);

    // 2c) softmax over last dim
    softmax_k<<<(Bsz * H * S) / 8, 256, 0, stream>>>(scores, S,
                                                     (long long)Bsz * H * S);

    // 2d) o[b,h] = attn @ v   (M=512,N=64,K=512; B is [K,N] = v rows)
    launch_gemm(true, dim3(HD / 64, S / 128, Bsz * H),
                scores, S, qkv + 2 * D /*v*/, 3 * D, attn_o, D,
                nullptr, nullptr, 0, S, HD, S, 1.f, 0, H,
                (long long)H * S * S, (long long)S * S,
                (long long)S * 3 * D, HD,
                (long long)S * D, HD, stream);

    // 2e) t1 = attn_o @ Wo^T + bo + h0 (residual)
    launch_gemm(false, dim3(D / 64, M / 128, 1), attn_o, D, Wo, D, t1, D,
                bo, h0, D, M, D, D, 1.f, 0, 1, 0, 0, 0, 0, 0, 0, stream);

    // 2f) h0 = LN1(t1)
    layernorm_k<<<M / 8, 256, 0, stream>>>(t1, ln1_g + (size_t)l * D,
                                           ln1_b + (size_t)l * D, h0, D, M);

    // 2g) ff1o = relu(h0 @ W1^T + b1)
    launch_gemm(false, dim3(FF / 64, M / 128, 1), h0, D, W1, D, ff1o, FF,
                b1, nullptr, 0, M, FF, D, 1.f, 1, 1, 0, 0, 0, 0, 0, 0, stream);

    // 2h) t1 = ff1o @ W2^T + b2 + h0
    launch_gemm(false, dim3(D / 64, M / 128, 1), ff1o, FF, W2, FF, t1, D,
                b2, h0, D, M, D, FF, 1.f, 0, 1, 0, 0, 0, 0, 0, 0, stream);

    // 2i) h0 = LN2(t1)
    layernorm_k<<<M / 8, 256, 0, stream>>>(t1, ln2_g + (size_t)l * D,
                                           ln2_b + (size_t)l * D, h0, D, M);
  }

  // 3) ESN input drive: uu = h0 @ W_in^T   ([B,S,R]; scan indexes [b,t,r])
  launch_gemm(false, dim3(R / 64, M / 128, 1), h0, D, W_in, D, uu, R,
              nullptr, nullptr, 0, M, R, D, 1.f, 0, 1, 0, 0, 0, 0, 0, 0, stream);

  // 4) W_res -> bf16 (stays hot in the 192MB L2 for all 512 scan steps)
  cvt_bf16_k<<<(R * R) / 256, 256, 0, stream>>>(W_res, wres_bf,
                                                (long long)R * R);

  // 5) sequential ESN scan + readout (single WG, 32 waves, LDS-resident state)
  esn_scan_k<<<1, 1024, 0, stream>>>(uu, wres_bf, ro_W, ro_b, (float*)d_out);
}
